// MultiHeadAttention_87462714015829
// MI455X (gfx1250) — matrix operand order repaired and checked by reading, not compiled, not run on hardware
//
#include <hip/hip_runtime.h>
#include <hip/hip_bf16.h>
#include <math.h>

#define BATCH  4
#define SEQ    2048
#define DMODEL 1024
#define NHEAD  8
#define DK     128
#define NEG_INF_F (-1e9f)

typedef __bf16 bf16_t;
typedef bf16_t v16bf __attribute__((ext_vector_type(16)));
typedef bf16_t v8bf  __attribute__((ext_vector_type(8)));
typedef float  v8f   __attribute__((ext_vector_type(8)));
typedef unsigned int u32x4 __attribute__((ext_vector_type(4)));
typedef unsigned int u32x8 __attribute__((ext_vector_type(8)));

union V16U { v16bf v; v8bf h[2]; };

__device__ __forceinline__ unsigned short f32_to_bf16_bits(float f) {
  union { float f; unsigned int u; } x; x.f = f;
  unsigned int u = x.u;
  u += 0x7FFFu + ((u >> 16) & 1u);   // round-to-nearest-even
  return (unsigned short)(u >> 16);
}
__device__ __forceinline__ unsigned int pack_bf16x2(float lo, float hi) {
  return (unsigned int)f32_to_bf16_bits(lo) |
         ((unsigned int)f32_to_bf16_bits(hi) << 16);
}
__device__ __forceinline__ v8bf ld_v8bf(const unsigned short* p) {
  return *(const v8bf*)p;            // 16B-aligned by construction
}
__device__ __forceinline__ v8f wmma_bf16(v16bf a, v16bf b, v8f c) {
  return __builtin_amdgcn_wmma_f32_16x16x32_bf16(
      false, a, false, b, (short)0, c, false, false);
}

// ---------------------------------------------------------------------------
// TDM: 2D bf16 tile load, global -> LDS, with hardware row padding.
// D# group0/group1 packed per CDNA5 ISA 8.3/8.4; groups 2/3 omitted (2D).
// pad: 16 DWORDs (one 32-elem bf16 row) then +8 DWORDs -> 48-elem LDS stride.
// ---------------------------------------------------------------------------
__device__ __forceinline__ void tdm_load_tile_bf16(
    unsigned int lds_byte_addr, const unsigned short* gptr,
    unsigned int tensor_d0, unsigned int tensor_d1,
    unsigned int stride_d0,             // in elements
    unsigned int tile_d0, unsigned int tile_d1) {
  unsigned long long ga = (unsigned long long)gptr;
  u32x4 g0;
  g0[0] = 1u;                                        // count=1, user D#
  g0[1] = lds_byte_addr;                             // lds_addr
  g0[2] = (unsigned int)ga;                          // global_addr[31:0]
  g0[3] = ((unsigned int)(ga >> 32) & 0x01FFFFFFu)   // global_addr[56:32]
          | 0x80000000u;                             // type=2 ("image")
  u32x8 g1;
  g1[0] = (1u << 16)      // data_size = 1 (2 bytes, bf16)
        | (1u << 20)      // pad_enable
        | (3u << 22)      // pad_interval code 3 = 16 DWORDs
        | (7u << 25);     // pad_amount  code 7 =  8 DWORDs
  g1[1] = (tensor_d0 & 0xFFFFu) << 16;               // tensor_dim0[15:0]
  g1[2] = (tensor_d0 >> 16) | ((tensor_d1 & 0xFFFFu) << 16);
  g1[3] = (tensor_d1 >> 16) | (tile_d0 << 16);       // tile_dim0
  g1[4] = tile_d1;                                   // tile_dim1 (tile_dim2=0)
  g1[5] = stride_d0;                                 // tensor_dim0_stride[31:0]
  g1[6] = 0u;                                        // stride hi / dim1_stride lo
  g1[7] = 0u;
  asm volatile("tensor_load_to_lds %0, %1" :: "s"(g0), "s"(g1) : "memory");
}

// ---------------------------------------------------------------------------
// Kernel 0a: fp32 -> bf16 copy of X (queries/keys/values), 8 elems/thread.
// ---------------------------------------------------------------------------
__global__ __launch_bounds__(256)
void cvt_x_kernel(const float* __restrict__ Xq, const float* __restrict__ Xk,
                  const float* __restrict__ Xv, unsigned short* __restrict__ dstAll) {
  const int mat = blockIdx.y;
  const float* src = (mat == 0) ? Xq : (mat == 1) ? Xk : Xv;
  unsigned short* dst = dstAll + (size_t)mat * BATCH * SEQ * DMODEL;
  const size_t i = ((size_t)blockIdx.x * 256 + threadIdx.x) * 8;
  float4 a = *(const float4*)&src[i];
  float4 b = *(const float4*)&src[i + 4];
  uint4 o;
  o.x = pack_bf16x2(a.x, a.y);
  o.y = pack_bf16x2(a.z, a.w);
  o.z = pack_bf16x2(b.x, b.y);
  o.w = pack_bf16x2(b.z, b.w);
  *(uint4*)&dst[i] = o;
}

// ---------------------------------------------------------------------------
// Kernel 0b: fp32 -> bf16 transpose of W: Wt[n][k] = W[k][n].  32x32 LDS tile.
// ---------------------------------------------------------------------------
__global__ __launch_bounds__(256)
void cvt_w_kernel(const float* __restrict__ Wq, const float* __restrict__ Wk,
                  const float* __restrict__ Wv, unsigned short* __restrict__ dstAll) {
  __shared__ float tile[32][33];
  const int mat = blockIdx.z;
  const float* W = (mat == 0) ? Wq : (mat == 1) ? Wk : Wv;
  unsigned short* Wt = dstAll + (size_t)mat * DMODEL * DMODEL;
  const int n0 = blockIdx.x * 32;
  const int k0 = blockIdx.y * 32;
  const int t = threadIdx.x;
  {
    int r = t >> 3, c = (t & 7) * 4;                 // read W[k0+r][n0+c..+3]
    float4 d = *(const float4*)&W[(size_t)(k0 + r) * DMODEL + n0 + c];
    tile[r][c] = d.x; tile[r][c + 1] = d.y; tile[r][c + 2] = d.z; tile[r][c + 3] = d.w;
  }
  __syncthreads();
  {
    int nr = t >> 3, kg = (t & 7) * 4;               // write Wt[n0+nr][k0+kg..+3]
    uint2 o;
    o.x = pack_bf16x2(tile[kg][nr],     tile[kg + 1][nr]);
    o.y = pack_bf16x2(tile[kg + 2][nr], tile[kg + 3][nr]);
    *(uint2*)&Wt[(size_t)(n0 + nr) * DMODEL + k0 + kg] = o;
  }
}

// ---------------------------------------------------------------------------
// Kernel 1: QKV projection GEMM, bf16 WMMA, TDM-staged LDS double buffering.
// grid = (64 row-tiles, 8 col-tiles, 3 matrices), 256 threads (8 waves).
// ---------------------------------------------------------------------------
#define KTILES (DMODEL / 32)
__global__ __launch_bounds__(256)
void qkv_proj_kernel(const unsigned short* __restrict__ Xbf,   // [3][8192][1024]
                     const unsigned short* __restrict__ Wtbf,  // [3][1024 n][1024 k]
                     const float* __restrict__ bq, const float* __restrict__ bk,
                     const float* __restrict__ bv,
                     unsigned short* __restrict__ wsQ,
                     unsigned short* __restrict__ wsK,
                     unsigned short* __restrict__ wsV) {
  constexpr int LDT = 48;  // padded LDS row stride (matches TDM pad codes)
  __shared__ __align__(16) unsigned short Abuf[2][128 * LDT];
  __shared__ __align__(16) unsigned short Bbuf[2][128 * LDT];

  const int tid  = threadIdx.x;
  const int lane = tid & 31;
  const int wid  = tid >> 5;
  const int wr   = wid & 3;
  const int wc   = wid >> 2;
  const int half = lane >> 4;
  const int nl   = lane & 15;

  const int rowBase = blockIdx.x * 128;
  const int colBase = blockIdx.y * 128;
  const int mat     = blockIdx.z;

  const unsigned short* X  = Xbf  + (size_t)mat * BATCH * SEQ * DMODEL;
  const unsigned short* Wt = Wtbf + (size_t)mat * DMODEL * DMODEL;
  const float* bias = (mat == 0) ? bq : (mat == 1) ? bk : bv;

  v8f acc[2][4];
  for (int mt = 0; mt < 2; ++mt)
    for (int nt = 0; nt < 4; ++nt)
      for (int r = 0; r < 8; ++r) acc[mt][nt][r] = 0.0f;

  // Prologue: fill both buffers (fills complete in issue order; TENSORcnt).
  if (wid == 0) {
    for (int p = 0; p < 2; ++p) {
      tdm_load_tile_bf16((unsigned int)(size_t)&Abuf[p][0],
                         X + (size_t)rowBase * DMODEL + p * 32,
                         DMODEL, BATCH * SEQ, DMODEL, 32, 128);
      tdm_load_tile_bf16((unsigned int)(size_t)&Bbuf[p][0],
                         Wt + (size_t)colBase * DMODEL + p * 32,
                         DMODEL, DMODEL, DMODEL, 32, 128);
    }
  }

  for (int k = 0; k < KTILES; ++k) {
    const int cur = k & 1;
    if (wid == 0) {   // fill k done when <= 2 tensor ops (fill k+1) outstanding
      if (k < KTILES - 1) __builtin_amdgcn_s_wait_tensorcnt(2);
      else                __builtin_amdgcn_s_wait_tensorcnt(0);
    }
    __syncthreads();   // publish TDM-written LDS to all waves

    V16U af[2], bfm[4];
    for (int mt = 0; mt < 2; ++mt) {
      int row = wr * 32 + mt * 16 + nl;
      int k0  = half * 8;
      af[mt].h[0] = ld_v8bf(&Abuf[cur][row * LDT + k0]);
      af[mt].h[1] = ld_v8bf(&Abuf[cur][row * LDT + k0 + 16]);
    }
    for (int nt = 0; nt < 4; ++nt) {
      int nrow = wc * 64 + nt * 16 + nl;
      int k0   = half * 8;
      bfm[nt].h[0] = ld_v8bf(&Bbuf[cur][nrow * LDT + k0]);
      bfm[nt].h[1] = ld_v8bf(&Bbuf[cur][nrow * LDT + k0 + 16]);
    }
    for (int mt = 0; mt < 2; ++mt)
      for (int nt = 0; nt < 4; ++nt)
        acc[mt][nt] = wmma_bf16(af[mt].v, bfm[nt].v, acc[mt][nt]);

    __syncthreads();   // all waves done reading buf[cur]
    if (wid == 0 && k + 2 < KTILES) {  // refill buf[cur] with tile k+2
      tdm_load_tile_bf16((unsigned int)(size_t)&Abuf[cur][0],
                         X + (size_t)rowBase * DMODEL + (k + 2) * 32,
                         DMODEL, BATCH * SEQ, DMODEL, 32, 128);
      tdm_load_tile_bf16((unsigned int)(size_t)&Bbuf[cur][0],
                         Wt + (size_t)colBase * DMODEL + (k + 2) * 32,
                         DMODEL, DMODEL, DMODEL, 32, 128);
    }
  }

  // Epilogue: bias, bf16 convert, scatter (Q,K: [B,H,S,dk]; V: [B,H,dk,S]).
  unsigned short* outp = (mat == 0) ? wsQ : (mat == 1) ? wsK : wsV;
  const bool isV = (mat == 2);
  for (int mt = 0; mt < 2; ++mt) {
    for (int nt = 0; nt < 4; ++nt) {
      int col = colBase + wc * 64 + nt * 16 + nl;
      int hh  = col >> 7;
      int f   = col & 127;
      float bb = bias[col];
      for (int r = 0; r < 8; ++r) {
        int g = rowBase + wr * 32 + mt * 16 + r + half * 8;
        int b = g >> 11;
        int s = g & (SEQ - 1);
        int base = (b * NHEAD + hh);
        int idx_qk = (base * SEQ + s) * DK + f;
        int idx_v  = (base * DK + f) * SEQ + s;
        outp[isV ? idx_v : idx_qk] = f32_to_bf16_bits(acc[mt][nt][r] + bb);
      }
    }
  }
}

// ---------------------------------------------------------------------------
// Kernel 2: causal flash attention.  One wave per 16-query tile, 32-key blocks.
// ---------------------------------------------------------------------------
__global__ __launch_bounds__(128)
void flash_attn_kernel(const unsigned short* __restrict__ wsQ,
                       const unsigned short* __restrict__ wsK,
                       const unsigned short* __restrict__ wsV,
                       float* __restrict__ wsO) {
  __shared__ __align__(16) unsigned short Plds[4][16 * 32];

  const int tid  = threadIdx.x;
  const int lane = tid & 31;
  const int wid  = tid >> 5;
  const int half = lane >> 4;
  const int nl   = lane & 15;

  const int qt = blockIdx.x * 4 + wid;
  const int q0 = qt * 16;
  const int h  = blockIdx.y;
  const int b  = blockIdx.z;

  const unsigned short* Qbh = wsQ + (size_t)((b * NHEAD + h) * SEQ) * DK;
  const unsigned short* Kbh = wsK + (size_t)((b * NHEAD + h) * SEQ) * DK;
  const unsigned short* Vbh = wsV + (size_t)((b * NHEAD + h) * DK) * SEQ;

  V16U qa[4];
  for (int c = 0; c < 4; ++c) {
    const unsigned short* p = Qbh + (size_t)(q0 + nl) * DK + c * 32 + half * 8;
    qa[c].h[0] = ld_v8bf(p);
    qa[c].h[1] = ld_v8bf(p + 16);
  }

  v8f o[8];
  for (int t = 0; t < 8; ++t)
    for (int r = 0; r < 8; ++r) o[t][r] = 0.0f;
  float mrow[8], lrow[8];
  for (int r = 0; r < 8; ++r) { mrow[r] = -3.0e38f; lrow[r] = 0.0f; }

  const float scale = 0.088388347648318447f;  // 1/sqrt(128)

  for (int kb = 0; kb < q0 + 16; kb += 32) {
    v8f sc[2];
    for (int t = 0; t < 2; ++t)
      for (int r = 0; r < 8; ++r) sc[t][r] = 0.0f;
    for (int c = 0; c < 4; ++c) {
      for (int t = 0; t < 2; ++t) {
        const unsigned short* p =
            Kbh + (size_t)(kb + t * 16 + nl) * DK + c * 32 + half * 8;
        V16U kf;
        kf.h[0] = ld_v8bf(p);
        kf.h[1] = ld_v8bf(p + 16);
        sc[t] = wmma_bf16(qa[c].v, kf.v, sc[t]);
      }
    }
    if (kb + 32 < SEQ) {
      __builtin_prefetch(Kbh + (size_t)(kb + 32 + nl) * DK, 0, 1);
      __builtin_prefetch(Vbh + (size_t)nl * SEQ + kb + 32, 0, 1);
    }
    for (int r = 0; r < 8; ++r) {
      int q = q0 + r + half * 8;
      float s0 = sc[0][r] * scale;
      float s1 = sc[1][r] * scale;
      if (kb + nl > q)      s0 = NEG_INF_F;
      if (kb + 16 + nl > q) s1 = NEG_INF_F;
      float mx = fmaxf(s0, s1);
      for (int off = 1; off < 16; off <<= 1)
        mx = fmaxf(mx, __shfl_xor(mx, off, 16));
      float mnew  = fmaxf(mrow[r], mx);
      float alpha = __expf(mrow[r] - mnew);
      float p0 = __expf(s0 - mnew);
      float p1 = __expf(s1 - mnew);
      float ps = p0 + p1;
      for (int off = 1; off < 16; off <<= 1)
        ps += __shfl_xor(ps, off, 16);
      mrow[r] = mnew;
      lrow[r] = lrow[r] * alpha + ps;
      for (int t = 0; t < 8; ++t) o[t][r] *= alpha;
      int M = r + half * 8;
      Plds[wid][M * 32 + nl]      = f32_to_bf16_bits(p0);
      Plds[wid][M * 32 + 16 + nl] = f32_to_bf16_bits(p1);
    }
    V16U pf;
    {
      const unsigned short* p = &Plds[wid][nl * 32 + half * 8];
      pf.h[0] = ld_v8bf(p);
      pf.h[1] = ld_v8bf(p + 16);
    }
    for (int t = 0; t < 8; ++t) {
      const unsigned short* p = Vbh + (size_t)(t * 16 + nl) * SEQ + kb + half * 8;
      V16U vf;
      vf.h[0] = ld_v8bf(p);
      vf.h[1] = ld_v8bf(p + 16);
      o[t] = wmma_bf16(pf.v, vf.v, o[t]);
    }
  }

  for (int r = 0; r < 8; ++r) {
    float inv = 1.0f / lrow[r];
    for (int t = 0; t < 8; ++t) o[t][r] *= inv;
  }
  for (int r = 0; r < 8; ++r) {
    int s = q0 + r + half * 8;
    float* dst = wsO + (size_t)(b * SEQ + s) * DMODEL + h * DK;
    for (int t = 0; t < 8; ++t) dst[t * 16 + nl] = o[t][r];
  }
}

// ---------------------------------------------------------------------------
// Kernel 3: residual + LayerNorm over D=1024.  One 256-thread WG per token.
// ---------------------------------------------------------------------------
__global__ __launch_bounds__(256)
void resid_ln_kernel(const float* __restrict__ attn,
                     const float* __restrict__ queries,
                     const float* __restrict__ gamma,
                     const float* __restrict__ beta,
                     float* __restrict__ out) {
  __shared__ float s1[256];
  __shared__ float s2[256];
  const int row = blockIdx.x;
  const int tid = threadIdx.x;
  const size_t base = (size_t)row * DMODEL;
  const int d0 = tid * 4;

  float4 a = *(const float4*)&attn[base + d0];
  float4 q = *(const float4*)&queries[base + d0];
  float r0 = a.x + q.x, r1 = a.y + q.y, r2 = a.z + q.z, r3 = a.w + q.w;
  s1[tid] = r0 + r1 + r2 + r3;
  s2[tid] = r0 * r0 + r1 * r1 + r2 * r2 + r3 * r3;
  __syncthreads();
  for (int off = 128; off > 0; off >>= 1) {
    if (tid < off) { s1[tid] += s1[tid + off]; s2[tid] += s2[tid + off]; }
    __syncthreads();
  }
  const float mu   = s1[0] * (1.0f / DMODEL);
  const float var  = s2[0] * (1.0f / DMODEL) - mu * mu;
  const float rstd = rsqrtf(var + 1e-6f);

  float4 g  = *(const float4*)&gamma[d0];
  float4 bt = *(const float4*)&beta[d0];
  float4 ov;
  ov.x = (r0 - mu) * rstd * g.x + bt.x;
  ov.y = (r1 - mu) * rstd * g.y + bt.y;
  ov.z = (r2 - mu) * rstd * g.z + bt.z;
  ov.w = (r3 - mu) * rstd * g.w + bt.w;
  *(float4*)&out[base + d0] = ov;
}

// ---------------------------------------------------------------------------
extern "C" void kernel_launch(void* const* d_in, const int* in_sizes, int n_in,
                              void* d_out, int out_size, void* d_ws, size_t ws_size,
                              hipStream_t stream) {
  (void)in_sizes; (void)n_in; (void)out_size; (void)ws_size;
  const float* Xq    = (const float*)d_in[0];
  const float* Xk    = (const float*)d_in[1];
  const float* Xv    = (const float*)d_in[2];
  const float* Wq    = (const float*)d_in[3];
  const float* bq    = (const float*)d_in[4];
  const float* Wk    = (const float*)d_in[5];
  const float* bk    = (const float*)d_in[6];
  const float* Wv    = (const float*)d_in[7];
  const float* bv    = (const float*)d_in[8];
  const float* gamma = (const float*)d_in[9];
  const float* beta  = (const float*)d_in[10];
  float* out = (float*)d_out;

  // Workspace map (stream-ordered reuse):
  //   [0,48MB)   : bf16 X (3x16MB) for kernels 0a/1; reused as fp32 attn-out
  //   [48,54MB)  : bf16 W^T (3x2MB)
  //   [54,102MB) : bf16 Q,K,V (3x16MB)
  char* ws = (char*)d_ws;
  unsigned short* Xbf  = (unsigned short*)(ws);
  unsigned short* Wtbf = (unsigned short*)(ws + (size_t)48 * 1024 * 1024);
  unsigned short* wsQ  = (unsigned short*)(ws + (size_t)54 * 1024 * 1024);
  unsigned short* wsK  = (unsigned short*)(ws + (size_t)70 * 1024 * 1024);
  unsigned short* wsV  = (unsigned short*)(ws + (size_t)86 * 1024 * 1024);
  float*          wsO  = (float*)(ws);  // overlays Xbf (dead after GEMM)

  dim3 g0a(BATCH * SEQ * DMODEL / (256 * 8), 3), b256(256);
  cvt_x_kernel<<<g0a, b256, 0, stream>>>(Xq, Xk, Xv, Xbf);

  dim3 g0b(DMODEL / 32, DMODEL / 32, 3);
  cvt_w_kernel<<<g0b, b256, 0, stream>>>(Wq, Wk, Wv, Wtbf);

  dim3 gA(BATCH * SEQ / 128, DMODEL / 128, 3);
  qkv_proj_kernel<<<gA, b256, 0, stream>>>(Xbf, Wtbf, bq, bk, bv, wsQ, wsK, wsV);

  dim3 gB(SEQ / 64, NHEAD, BATCH), bB(128);
  flash_attn_kernel<<<gB, bB, 0, stream>>>(wsQ, wsK, wsV, wsO);

  dim3 gC(BATCH * SEQ);
  resid_ln_kernel<<<gC, b256, 0, stream>>>(wsO, Xq, gamma, beta, out);
}
